// MambaBlock_34565896798335
// MI455X (gfx1250) — matrix operand order repaired and checked by reading, not compiled, not run on hardware
//
#include <hip/hip_runtime.h>
#include <hip/hip_bf16.h>
#include <math.h>

// ---------------- problem constants (fixed by the reference) ----------------
#define DIMC   2048
#define INNERC 4096
#define LC     2048
#define BC     2
#define NST    16
#define DTR    128
#define ML     (BC * LC)          // 4096 rows in every GEMM
#define XPN    (DTR + 2 * NST)    // 160

typedef __attribute__((ext_vector_type(4)))  float  v4f;
typedef __attribute__((ext_vector_type(8)))  float  v8f;
typedef __attribute__((ext_vector_type(8)))  __bf16 v8bf;
typedef __attribute__((ext_vector_type(16))) __bf16 v16bf;

// ---------------- generic NT WMMA GEMM: C[m,n] = sum_k A[m,k]*B[n,k] --------
// fp32 inputs split into bf16 hi/lo during LDS staging; per K=32 chunk:
//   acc += Ahi*Bhi + Ahi*Blo + Alo*Bhi   (3x v_wmma_f32_16x16x32_bf16)
// mode 0: plain store; mode 1: C = softplus(C + bias[n])
#define BM 128
#define BN 128
#define BK 32
#define LDSA 40   // bf16 row stride: 80B rows -> 16B-aligned, conflict-free b128 frags

__device__ __forceinline__ unsigned bf16rne(float x) {
    unsigned u = __float_as_uint(x);
    return (u + 0x7FFFu + ((u >> 16) & 1u)) >> 16;   // round-to-nearest-even bf16 bits
}

// split 4 floats into packed bf16 hi and lo pairs
__device__ __forceinline__ void split4(v4f v, uint2& hi, uint2& lo) {
    float f[4] = {v.x, v.y, v.z, v.w};
    unsigned h[4], l[4];
#pragma unroll
    for (int i = 0; i < 4; ++i) {
        h[i] = bf16rne(f[i]);
        float r = f[i] - __uint_as_float(h[i] << 16);  // exact residual
        l[i] = bf16rne(r);
    }
    hi.x = h[0] | (h[1] << 16); hi.y = h[2] | (h[3] << 16);
    lo.x = l[0] | (l[1] << 16); lo.y = l[2] | (l[3] << 16);
}

// load a 16-element bf16 fragment from two 16B LDS chunks (p and p+off)
__device__ __forceinline__ v16bf frag16(const __bf16* p, int off) {
    v8bf a = *(const v8bf*)(p);
    v8bf b = *(const v8bf*)(p + off);
    return __builtin_shufflevector(a, b, 0,1,2,3,4,5,6,7,8,9,10,11,12,13,14,15);
}

__global__ void __launch_bounds__(256)
wmma_gemm_nt(const float* __restrict__ A, int lda,
             const float* __restrict__ B, int ldb,
             float* __restrict__ C, int ldc,
             int M, int N, int K,
             const float* __restrict__ bias, int mode)
{
    __shared__ __bf16 Ahi[BM * LDSA];
    __shared__ __bf16 Alo[BM * LDSA];
    __shared__ __bf16 Bhi[BN * LDSA];
    __shared__ __bf16 Blo[BN * LDSA];

    const int tid  = threadIdx.x;
    const int wave = tid >> 5;        // 0..7 (wave32)
    const int lane = tid & 31;
    const int half = lane >> 4;       // lane-half per ISA fragment layout
    const int lr   = lane & 15;
    const int m0   = blockIdx.y * BM;
    const int n0   = blockIdx.x * BN;

    const int ldrow = tid >> 3;       // 0..31
    const int ldcol = (tid & 7) << 2; // 0,4,...,28

    v8f acc[8] = {};                  // 8 x (16x16 f32 tile) accumulators

    for (int kk = 0; kk < K; kk += BK) {
        __syncthreads();              // previous compute done before overwrite
#pragma unroll
        for (int i = 0; i < 4; ++i) {
            int row = ldrow + i * 32;
            int gm  = m0 + row;
            v4f va = {};
            if (gm < M) va = *(const v4f*)(A + (size_t)gm * lda + kk + ldcol);
            uint2 hi, lo;
            split4(va, hi, lo);
            *(uint2*)(&Ahi[row * LDSA + ldcol]) = hi;
            *(uint2*)(&Alo[row * LDSA + ldcol]) = lo;

            int gn  = n0 + row;
            v4f vb = {};
            if (gn < N) vb = *(const v4f*)(B + (size_t)gn * ldb + kk + ldcol);
            split4(vb, hi, lo);
            *(uint2*)(&Bhi[row * LDSA + ldcol]) = hi;
            *(uint2*)(&Blo[row * LDSA + ldcol]) = lo;
        }
        // pull next K-tile toward L2 (gfx1250 global_prefetch path)
        if (kk + BK < K) {
            int gm = m0 + ldrow;
            if (gm < M) __builtin_prefetch(A + (size_t)gm * lda + kk + BK + ldcol, 0, 1);
            int gn = n0 + ldrow;
            if (gn < N) __builtin_prefetch(B + (size_t)gn * ldb + kk + BK + ldcol, 0, 1);
        }
        __syncthreads();

        // A 16x32 bf16 frag: lanes0-15 hold K 0-7 & 16-23, lanes16-31 K 8-15 & 24-31
        const __bf16* ap = &Ahi[(wave * 16 + lr) * LDSA + 8 * half];
        const int aoff = (int)(Alo - Ahi);           // hi->lo tile offset (elements)
        v16bf ahi = frag16(ap, 16);
        v16bf alo = frag16(ap + aoff, 16);
#pragma unroll
        for (int j = 0; j < 8; ++j) {
            // B 32x16 bf16 frag: lanes0-15 hold K 0-7 & 16-23, lanes16-31 K 8-15 & 24-31
            const __bf16* bp = &Bhi[(j * 16 + lr) * LDSA + 8 * half];
            const int boff = (int)(Blo - Bhi);
            v16bf bhi = frag16(bp, 16);
            v16bf blo = frag16(bp + boff, 16);
            acc[j] = __builtin_amdgcn_wmma_f32_16x16x32_bf16(
                false, ahi, false, bhi, (short)0, acc[j], false, false);
            acc[j] = __builtin_amdgcn_wmma_f32_16x16x32_bf16(
                false, ahi, false, blo, (short)0, acc[j], false, false);
            acc[j] = __builtin_amdgcn_wmma_f32_16x16x32_bf16(
                false, alo, false, bhi, (short)0, acc[j], false, false);
        }
    }

    // C/D layout: lanes0-15: vgpr r -> M=r, N=lane ; lanes16-31: M=r+8, N=lane-16
    const int rbase = m0 + wave * 16 + half * 8;
#pragma unroll
    for (int j = 0; j < 8; ++j) {
        int col = n0 + j * 16 + lr;
        if (col < N) {
            float bv = (mode == 1) ? bias[col] : 0.0f;
#pragma unroll
            for (int r = 0; r < 8; ++r) {
                int row = rbase + r;
                if (row < M) {
                    float v = acc[j][r];
                    if (mode == 1) {
                        v += bv;
                        v = (v > 20.0f) ? v : log1pf(__expf(v));  // softplus
                    }
                    C[(size_t)row * ldc + col] = v;
                }
            }
        }
    }
}

// ------------- depthwise causal conv1d (K=4) + bias + SiLU ------------------
// reads u-half of xz (cols [0,INNER) of width 2*INNER), writes u buffer
__global__ void __launch_bounds__(256)
conv_silu(const float* __restrict__ xz, const float* __restrict__ w,
          const float* __restrict__ bias, float* __restrict__ u)
{
    size_t idx = (size_t)blockIdx.x * 256 + threadIdx.x;  // over ML*INNERC exactly
    int d  = (int)(idx & (INNERC - 1));
    int bl = (int)(idx >> 12);            // INNERC == 4096 == 2^12
    int l  = bl & (LC - 1);               // LC power of two; bl = b*LC + l
    float acc = bias[d];
    const float* wd = w + d * 4;
#pragma unroll
    for (int k = 0; k < 4; ++k) {
        int lk = l - 3 + k;               // causal left pad of K-1
        if (lk >= 0)
            acc += xz[(size_t)(bl - 3 + k) * (2 * INNERC) + d] * wd[k];
    }
    u[idx] = acc / (1.0f + __expf(-acc)); // SiLU
}

// ------------- selective scan, fused with skip (u*D) and SiLU gate ----------
// one thread per (b,d) channel: 16-state recurrence in registers over L steps.
// B_t/C_t (block-uniform) broadcast via LDS. Gated output overwrites the
// u-half of xz (free after conv), read later by the out_proj GEMM.
__global__ void __launch_bounds__(256)
scan_fused(const float* __restrict__ delta, const float* __restrict__ u,
           float* __restrict__ xz, const float* __restrict__ x_dbl,
           const float* __restrict__ A_log, const float* __restrict__ Dvec)
{
    const int d = blockIdx.x * 256 + threadIdx.x;  // channel
    const int b = blockIdx.y;
    __shared__ float sBC[2 * NST];

    float a[NST], h[NST];
#pragma unroll
    for (int n = 0; n < NST; ++n) {
        a[n] = -__expf(A_log[(size_t)d * NST + n]);
        h[n] = 0.0f;
    }
    const float Dd = Dvec[d];

    for (int l = 0; l < LC; ++l) {
        size_t base = (size_t)b * LC + l;
        if (threadIdx.x < 2 * NST)
            sBC[threadIdx.x] = x_dbl[base * XPN + DTR + threadIdx.x];
        __syncthreads();

        float dt = delta[base * INNERC + d];
        float ut = u[base * INNERC + d];
        float zt = xz[base * (2 * INNERC) + INNERC + d];   // z half
        float du = dt * ut;
        float y  = 0.0f;
#pragma unroll
        for (int n = 0; n < NST; ++n) {
            float dA = __expf(dt * a[n]);
            h[n] = dA * h[n] + du * sBC[n];
            y   += h[n] * sBC[NST + n];
        }
        y = (y + ut * Dd) * (zt / (1.0f + __expf(-zt)));   // skip + SiLU gate
        xz[base * (2 * INNERC) + d] = y;                    // reuse u-half
        __syncthreads();
    }
}

// ---------------------------------------------------------------------------
extern "C" void kernel_launch(void* const* d_in, const int* in_sizes, int n_in,
                              void* d_out, int out_size, void* d_ws, size_t ws_size,
                              hipStream_t stream)
{
    const float* x          = (const float*)d_in[0];   // (B,L,DIM)
    const float* in_proj_w  = (const float*)d_in[1];   // (2*INNER, DIM)
    const float* conv_w     = (const float*)d_in[2];   // (INNER,1,4)
    const float* conv_b     = (const float*)d_in[3];   // (INNER,)
    const float* x_proj_w   = (const float*)d_in[4];   // (160, INNER)
    const float* dt_proj_w  = (const float*)d_in[5];   // (INNER, 128)
    const float* dt_proj_b  = (const float*)d_in[6];   // (INNER,)
    const float* A_log      = (const float*)d_in[7];   // (INNER, 16)
    const float* Dvec       = (const float*)d_in[8];   // (INNER,)
    const float* out_proj_w = (const float*)d_in[9];   // (DIM, INNER)
    float* out = (float*)d_out;                        // (B,L,DIM) fp32

    // workspace layout (floats)
    float* ws    = (float*)d_ws;
    float* xz    = ws;                                  // ML * 2*INNER   (128 MB)
    float* u     = xz    + (size_t)ML * 2 * INNERC;     // ML * INNER     ( 64 MB)
    float* x_dbl = u     + (size_t)ML * INNERC;         // ML * 160       (2.5 MB)
    float* delta = x_dbl + (size_t)ML * XPN;            // ML * INNER     ( 64 MB)

    dim3 blk(256);

    // 1) in_proj: xz = x @ in_proj_w^T      (4096 x 8192 x 2048)
    wmma_gemm_nt<<<dim3((2 * INNERC) / BN, ML / BM), blk, 0, stream>>>(
        x, DIMC, in_proj_w, DIMC, xz, 2 * INNERC, ML, 2 * INNERC, DIMC, nullptr, 0);

    // 2) depthwise causal conv + bias + SiLU on u-half
    conv_silu<<<(size_t)ML * INNERC / 256, blk, 0, stream>>>(xz, conv_w, conv_b, u);

    // 3) x_proj: x_dbl = u @ x_proj_w^T     (4096 x 160 x 4096)
    wmma_gemm_nt<<<dim3((XPN + BN - 1) / BN, ML / BM), blk, 0, stream>>>(
        u, INNERC, x_proj_w, INNERC, x_dbl, XPN, ML, XPN, INNERC, nullptr, 0);

    // 4) dt_proj + softplus: delta = softplus(x_dbl[:, :128] @ dt_proj_w^T + b)
    wmma_gemm_nt<<<dim3(INNERC / BN, ML / BM), blk, 0, stream>>>(
        x_dbl, XPN, dt_proj_w, DTR, delta, INNERC, ML, INNERC, DTR, dt_proj_b, 1);

    // 5) selective scan + skip + gate  -> written into u-half of xz
    scan_fused<<<dim3(INNERC / 256, BC), blk, 0, stream>>>(
        delta, u, xz, x_dbl, A_log, Dvec);

    // 6) out_proj: out = y @ out_proj_w^T   (4096 x 2048 x 4096)
    wmma_gemm_nt<<<dim3(DIMC / BN, ML / BM), blk, 0, stream>>>(
        xz, 2 * INNERC, out_proj_w, INNERC, out, DIMC, ML, DIMC, INNERC, nullptr, 0);
}
